// HippocampalPredictiveRNN_70308614636251
// MI455X (gfx1250) — matrix operand order repaired and checked by reading, not compiled, not run on hardware
//
#include <hip/hip_runtime.h>
#include <hip/hip_bf16.h>

// ---------------------------------------------------------------------------
// Leaky threshold-linear RNN scan on MI455X (gfx1250, wave32, WMMA).
// 4 persistent workgroups x 512 threads (16 wave32s); each WG owns 16 batch
// rows. h_t lives on-chip (bf16 A-copy in LDS + f32 master in registers).
// Half of W_rec (bf16) is resident in the WGP's 320KB LDS; the other half
// plus W_out stream from L2 each step (laundered offset defeats LICM so the
// compiler can't hoist+spill them to scratch). W_in fragments (48 VGPRs) are
// intentionally left hoistable -> register-resident across the whole scan.
// ---------------------------------------------------------------------------

typedef __bf16 bf16_t;
typedef __attribute__((ext_vector_type(16))) __bf16 v16bf;
typedef __attribute__((ext_vector_type(8)))  __bf16 v8bf;
typedef __attribute__((ext_vector_type(8)))  float  v8f;

#define B_    64
#define T_    1024
#define OBS_  76
#define ACT_  3
#define D_    79      // OBS + ACT
#define H_    500
#define HP    512     // H padded to WMMA granularity
#define DP    96      // D padded to multiple of 32
#define OPAD  80      // OBS padded to multiple of 16
#define KLDS  256     // K-range of W_rec kept resident in LDS (bf16, 256KB)
#define WL_STRIDE 264 // LDS row stride for W_rec slice: bank-stride 4, 16B-aligned
#define HB_STRIDE 520 // LDS row stride for h (bf16)
#define XB_STRIDE 104 // LDS row stride for x_t (bf16)

#define WLDS_ELEMS (HP * WL_STRIDE)
#define HB_OFF     (WLDS_ELEMS)
#define XB_OFF     (HB_OFF + 16 * HB_STRIDE)
#define SMEM_ELEMS (XB_OFF + 16 * XB_STRIDE)
#define SMEM_BYTES (SMEM_ELEMS * 2)   // ~290 KB of the WGP's 320 KB LDS

// --- prep: quantize weights to zero-padded bf16 in workspace ----------------
__global__ __launch_bounds__(256) void prep_weights(
    const float* __restrict__ W_in, const float* __restrict__ W_rec,
    const float* __restrict__ mask, const float* __restrict__ W_out,
    bf16_t* __restrict__ Wrec_b, bf16_t* __restrict__ Win_b,
    bf16_t* __restrict__ Wout_b)
{
  int idx = blockIdx.x * 256 + threadIdx.x;
  if (idx < HP * HP) {                       // W_eff = W_rec * mask (REC_SCALE==1)
    int j = idx >> 9, k = idx & (HP - 1);
    float v = (j < H_ && k < H_) ? W_rec[j * H_ + k] * mask[j * H_ + k] : 0.f;
    Wrec_b[idx] = (bf16_t)v;
  }
  if (idx < HP * DP) {                       // W_in rows padded D_->DP
    int j = idx / DP, d = idx - (idx / DP) * DP;
    float v = (j < H_ && d < D_) ? W_in[j * D_ + d] : 0.f;
    Win_b[idx] = (bf16_t)v;
  }
  if (idx < OPAD * HP) {                     // W_out rows padded H_->HP
    int o = idx >> 9, k = idx & (HP - 1);
    float v = (o < OBS_ && k < H_) ? W_out[o * H_ + k] : 0.f;
    Wout_b[idx] = (bf16_t)v;
  }
}

// A(16x32 bf16) per-lane: e0..7 = A[m][ab..ab+7], e8..15 = A[m][ab+8..ab+15]
__device__ __forceinline__ v16bf make_a_frag(const bf16_t* row, int ab) {
  v8bf lo = *(const v8bf*)(row + ab);
  v8bf hi = *(const v8bf*)(row + ab + 8);
  return __builtin_shufflevector(lo, hi, 0, 1, 2, 3, 4, 5, 6, 7,
                                         8, 9, 10, 11, 12, 13, 14, 15);
}

// B(32x16 bf16) per-lane: 16 contiguous bf16 (two 16B LDS loads; rows are only
// 16B-aligned, so a single 32B load is not legal here).
__device__ __forceinline__ v16bf make_b_frag_lds(const bf16_t* p) {
  v8bf lo = *(const v8bf*)p;
  v8bf hi = *(const v8bf*)(p + 8);
  return __builtin_shufflevector(lo, hi, 0, 1, 2, 3, 4, 5, 6, 7,
                                         8, 9, 10, 11, 12, 13, 14, 15);
}

// --- persistent fused scan --------------------------------------------------
__global__ __launch_bounds__(512, 1) void rnn_scan(
    const float* __restrict__ obs, const float* __restrict__ act,
    const float* __restrict__ h0,  const float* __restrict__ noise,
    const float* __restrict__ bvec, const float* __restrict__ bout,
    const bf16_t* __restrict__ Wrec_b, const bf16_t* __restrict__ Win_b,
    const bf16_t* __restrict__ Wout_b, float* __restrict__ out)
{
  extern __shared__ __align__(16) bf16_t smem[];
  bf16_t* wl = smem;            // W_rec[:, 0:KLDS] resident slice [512][WL_STRIDE]
  bf16_t* hb = smem + HB_OFF;   // h_t  [16][HB_STRIDE]
  bf16_t* xb = smem + XB_OFF;   // x_t  [16][XB_STRIDE]

  const int tid  = threadIdx.x;
  const int wave = tid >> 5;      // 0..15
  const int lane = tid & 31;
  const int l15  = lane & 15;
  const int lhi  = lane >> 4;     // half-wave selector in fragment layouts
  const int b0   = blockIdx.x * 16;

  // stage the LDS-resident half of W_rec (once per launch)
  for (int i = tid; i < HP * (KLDS / 8); i += 512) {
    int j = i / (KLDS / 8);
    int c = (i - j * (KLDS / 8)) * 8;
    *(v8bf*)&wl[j * WL_STRIDE + c] = *(const v8bf*)&Wrec_b[(size_t)j * HP + c];
  }
  // init h in LDS from h0 (pad columns 500..511 = 0)
  for (int i = tid; i < 16 * HP; i += 512) {
    int m = i >> 9, k = i & (HP - 1);
    hb[m * HB_STRIDE + k] = (bf16_t)((k < H_) ? h0[(b0 + m) * H_ + k] : 0.f);
  }

  // f32 master copy of this lane's h cells: column n, rows m = lhi*8 + r
  float hold[2][8];
#pragma unroll
  for (int u = 0; u < 2; ++u) {
    int n = (wave * 2 + u) * 16 + l15;
#pragma unroll
    for (int r = 0; r < 8; ++r) {
      int m = lhi * 8 + r;
      hold[u][r] = (n < H_) ? h0[(b0 + m) * H_ + n] : 0.f;
    }
  }

  const v8f vzero = {0.f, 0.f, 0.f, 0.f, 0.f, 0.f, 0.f, 0.f};
  int tz = 0;  // laundered zero: re-defined every iteration to block LICM

  for (int t = 0; t < T_; ++t) {
    // ---- stage x_t = [obs|act] (bf16, zero-padded to DP) ----
    for (int i = tid; i < 16 * DP; i += 512) {
      int m = i / DP, d = i - m * DP;
      float v = 0.f;
      if (d < OBS_)    v = obs[((size_t)(b0 + m) * T_ + t) * OBS_ + d];
      else if (d < D_) v = act[((size_t)(b0 + m) * T_ + t) * ACT_ + (d - OBS_)];
      xb[m * XB_STRIDE + d] = (bf16_t)v;
    }
    if (t + 1 < T_) {  // warm L2 for next step's noise (global_prefetch_b8)
      __builtin_prefetch(&noise[((size_t)(t + 1) * B_ + b0 + l15) * H_ + wave * 32], 0, 1);
    }
    asm volatile("" : "+v"(tz));  // opaque 0: stops hoist+scratch-spill of W loads
    __syncthreads();              // xb and hb(t-1) visible

    v8f acc[2];
    acc[0] = vzero; acc[1] = vzero;

    // ---- input projection: pre += x_t @ W_in.T (K = 96).
    // No tz here on purpose: 6 fragments/wave -> compiler keeps W_in in VGPRs.
#pragma unroll
    for (int kk = 0; kk < DP; kk += 32) {
      v16bf afrag = make_a_frag(xb + l15 * XB_STRIDE, kk + lhi * 16);
#pragma unroll
      for (int u = 0; u < 2; ++u) {
        int j = (wave * 2 + u) * 16 + l15;          // B column n -> W_in row j
        v16bf bfrag = *(const v16bf*)(Win_b + (size_t)j * DP + kk + lhi * 16);
        acc[u] = __builtin_amdgcn_wmma_f32_16x16x32_bf16(
            false, afrag, false, bfrag, (short)0, acc[u], false, false);
      }
    }
    // ---- recurrence, K in [0, KLDS): B from LDS-resident W_rec slice ----
    for (int kk = 0; kk < KLDS; kk += 32) {
      v16bf afrag = make_a_frag(hb + l15 * HB_STRIDE, kk + lhi * 16);
#pragma unroll
      for (int u = 0; u < 2; ++u) {
        int j = (wave * 2 + u) * 16 + l15;
        v16bf bfrag = make_b_frag_lds(wl + j * WL_STRIDE + kk + lhi * 16 + tz);
        acc[u] = __builtin_amdgcn_wmma_f32_16x16x32_bf16(
            false, afrag, false, bfrag, (short)0, acc[u], false, false);
      }
    }
    // ---- recurrence, K in [KLDS, HP): B streamed from L2 ----
    for (int kk = KLDS; kk < HP; kk += 32) {
      v16bf afrag = make_a_frag(hb + l15 * HB_STRIDE, kk + lhi * 16);
#pragma unroll
      for (int u = 0; u < 2; ++u) {
        int j = (wave * 2 + u) * 16 + l15;
        v16bf bfrag = *(const v16bf*)(Wrec_b + (size_t)j * HP + kk + lhi * 16 + tz);
        acc[u] = __builtin_amdgcn_wmma_f32_16x16x32_bf16(
            false, afrag, false, bfrag, (short)0, acc[u], false, false);
      }
    }

    // ---- pointwise: h = 0.5*h + 0.5*relu(pre + b) + 0.03*noise ----
#pragma unroll
    for (int u = 0; u < 2; ++u) {
      int n = (wave * 2 + u) * 16 + l15;
      float bj = (n < H_) ? bvec[n] : 0.f;
#pragma unroll
      for (int r = 0; r < 8; ++r) {
        int m = lhi * 8 + r;
        float pre = acc[u][r] + bj;
        float nz = (n < H_) ? noise[((size_t)t * B_ + (b0 + m)) * H_ + n] : 0.f;
        hold[u][r] = 0.5f * hold[u][r] + 0.5f * fmaxf(pre, 0.f) + 0.03f * nz;
      }
    }
    __syncthreads();   // everyone's reads of hb(t-1) are done
#pragma unroll
    for (int u = 0; u < 2; ++u) {
      int n = (wave * 2 + u) * 16 + l15;
#pragma unroll
      for (int r = 0; r < 8; ++r) hb[(lhi * 8 + r) * HB_STRIDE + n] = (bf16_t)hold[u][r];
    }
    __syncthreads();   // hb = h_t

    // ---- readout: preds[:,t,:] = h_t @ W_out.T + b_out (waves 0..4) ----
    if (wave < 5) {
      v8f oacc = vzero;
      for (int kk = 0; kk < HP; kk += 32) {
        v16bf afrag = make_a_frag(hb + l15 * HB_STRIDE, kk + lhi * 16);
        int o = wave * 16 + l15;
        v16bf bfrag = *(const v16bf*)(Wout_b + (size_t)o * HP + kk + lhi * 16 + tz);
        oacc = __builtin_amdgcn_wmma_f32_16x16x32_bf16(
            false, afrag, false, bfrag, (short)0, oacc, false, false);
      }
      int o = wave * 16 + l15;
      if (o < OBS_) {
        float bo = bout[o];
#pragma unroll
        for (int r = 0; r < 8; ++r) {
          int m = lhi * 8 + r;
          out[((size_t)(b0 + m) * T_ + t) * OBS_ + o] = oacc[r] + bo;
        }
      }
    }
    // next iteration's first __syncthreads orders these hb reads against
    // the following step's hb update.
  }
}

// ---------------------------------------------------------------------------
extern "C" void kernel_launch(void* const* d_in, const int* in_sizes, int n_in,
                              void* d_out, int out_size, void* d_ws, size_t ws_size,
                              hipStream_t stream)
{
  (void)in_sizes; (void)n_in; (void)out_size; (void)ws_size;
  const float* obs   = (const float*)d_in[0];
  const float* act   = (const float*)d_in[1];
  const float* h0    = (const float*)d_in[2];
  const float* noise = (const float*)d_in[3];
  const float* W_in  = (const float*)d_in[4];
  const float* W_rec = (const float*)d_in[5];
  const float* mask  = (const float*)d_in[6];
  const float* bvec  = (const float*)d_in[7];
  const float* W_out = (const float*)d_in[8];
  const float* bout  = (const float*)d_in[9];

  // workspace: bf16 weights, ~705 KB total
  bf16_t* Wrec_b = (bf16_t*)d_ws;                  // 512*512
  bf16_t* Win_b  = Wrec_b + (size_t)HP * HP;       // 512*96
  bf16_t* Wout_b = Win_b + (size_t)HP * DP;        // 80*512

  prep_weights<<<dim3((HP * HP + 255) / 256), dim3(256), 0, stream>>>(
      W_in, W_rec, mask, W_out, Wrec_b, Win_b, Wout_b);
  rnn_scan<<<dim3(B_ / 16), dim3(512), SMEM_BYTES, stream>>>(
      obs, act, h0, noise, bvec, bout, Wrec_b, Win_b, Wout_b, (float*)d_out);
}
